// LSTM_48601849922171
// MI455X (gfx1250) — matrix operand order repaired and checked by reading, not compiled, not run on hardware
//
#include <hip/hip_runtime.h>
#include <hip/hip_bf16.h>

// ---------------------------------------------------------------------------
// 2-layer LSTM (B=8, T=2048, D=256, H=512), bf16 WMMA compute, fp32 state.
//   Phase 1: cvt fp32->bf16 (x, all weight matrices)
//   Phase 2: wx = A @ W^T + b, wave computes 16x64 strip (A reuse, 4
//            independent v_wmma_f32_16x16x32_bf16 per K-step)
//   Phase 3: recurrent scan on a 16-workgroup cluster, one WGP per 32
//            h-columns. Recurrent weights DMA'd into LDS once via the
//            Tensor Data Mover (tensor_load_to_lds + s_wait_tensorcnt);
//            h(t-1) broadcast to every WGP's LDS per step with the cluster
//            multicast async load (cluster_load_async_to_lds_b128 +
//            s_wait_asynccnt); cluster barriers for step sync.
// Workspace use: ~167 MB.
// ---------------------------------------------------------------------------

typedef __attribute__((ext_vector_type(16))) __bf16       v16bf;
typedef __attribute__((ext_vector_type(8)))  float        v8f;
typedef __attribute__((ext_vector_type(4))) unsigned int  v4u;
typedef __attribute__((ext_vector_type(8)))  int          v8i;
typedef __attribute__((ext_vector_type(4)))  int          v4i;

union V16 { v16bf v; v4u u[2]; };

#if defined(__AMDGCN__) && __has_builtin(__builtin_amdgcn_tensor_load_to_lds)
#define USE_TDM 1
#else
#define USE_TDM 0
#endif

// Cluster multicast async-load staging of h(t-1) into LDS (inline asm).
#if defined(__AMDGCN__)
#define USE_ASYNC_STAGE 1
#else
#define USE_ASYNC_STAGE 0
#endif

__device__ __forceinline__ unsigned short f2bf(float f) {
  unsigned int u = __builtin_bit_cast(unsigned int, f);
  return (unsigned short)((u + 0x7FFFu + ((u >> 16) & 1u)) >> 16);
}
__device__ __forceinline__ float sigmoidf_(float x) {
  return 1.0f / (1.0f + __expf(-x));
}

__global__ void cvt_f32_bf16(const float* __restrict__ in,
                             unsigned short* __restrict__ out, int n) {
  int i = blockIdx.x * blockDim.x + threadIdx.x;
  if (i < n) out[i] = f2bf(in[i]);
}

// C[m,n] = sum_k A[m,k]*Bm[n,k] + bias[n].  A:[M,K] bf16, Bm:[N,K] bf16.
// One wave computes a 16x64 strip: A loaded once per K-step, 4 independent
// WMMAs -> good co-execution pipelining.  8 waves / block.
__global__ void __launch_bounds__(256)
wmma_gemm_bias(const unsigned short* __restrict__ A,
               const unsigned short* __restrict__ Bm,
               const float* __restrict__ bias,
               float* __restrict__ C, int M, int N, int K) {
  const int wave = threadIdx.x >> 5;
  const int lane = threadIdx.x & 31;
  const int hi   = lane >> 4;
  const int nl   = lane & 15;
  const int gN   = N >> 6;                 // groups of 4 N-tiles
  const int tg   = blockIdx.x * 8 + wave;
  const int mt   = tg / gN;
  const int ng   = tg - mt * gN;
  if (mt * 16 >= M) return;                // wave-uniform

  const unsigned short* arow = A + (size_t)(mt * 16 + nl) * K;
  const unsigned short* br[4];
  #pragma unroll
  for (int j = 0; j < 4; ++j)
    br[j] = Bm + (size_t)(ng * 64 + j * 16 + nl) * K;

  v8f acc[4] = {};
  for (int k = 0; k < K; k += 32) {
    V16 a;
    a.u[0] = *(const v4u*)(arow + k + hi * 8);
    a.u[1] = *(const v4u*)(arow + k + 16 + hi * 8);
    #pragma unroll
    for (int j = 0; j < 4; ++j) {
      V16 b;
      b.u[0] = *(const v4u*)(br[j] + k + hi * 8);
      b.u[1] = *(const v4u*)(br[j] + k + 16 + hi * 8);
      acc[j] = __builtin_amdgcn_wmma_f32_16x16x32_bf16(false, a.v, false, b.v,
                                                       (short)0, acc[j], false, false);
    }
  }
  #pragma unroll
  for (int j = 0; j < 4; ++j) {
    const float bv = bias[ng * 64 + j * 16 + nl];
    float* cb = C + (size_t)(mt * 16 + hi * 8) * N + ng * 64 + j * 16 + nl;
    #pragma unroll
    for (int r = 0; r < 8; ++r) cb[(size_t)r * N] = acc[j][r] + bv;
  }
}

#define LSTM_H    512
#define LSTM_G    2048
#define SB_STRIDE 520   // 512 + 8 bf16 pad -> LDS rows start 4 banks apart

// 16-block cluster, 256 threads (8 waves) per block. Block g owns h-columns
// [g*32, g*32+32): wave w -> gate q=w>>1, half=w&1, 16x16 gate tile at
// N = q*512 + g*32 + half*16.  K=512 -> 16 WMMAs / wave / step, 2 acc chains.
__global__ void __launch_bounds__(256)
lstm_scan(const float* __restrict__ wx,            // [8*T, 2048] f32 (x-proj + bias)
          const unsigned short* __restrict__ uW,   // [2048, 512] bf16
          const float* __restrict__ ub,            // [2048]
          unsigned short* __restrict__ hcast,      // [16, 512] bf16 h broadcast
          unsigned short* __restrict__ hseq_bf,    // [8*T, 512] bf16 or null
          float* __restrict__ hseq_f32,            // [8*T, 512] f32 or null
          float* __restrict__ h_last,              // [8, 512]
          float* __restrict__ c_last,              // [8, 512]
          int T) {
  __shared__ __align__(16) unsigned short sB[8 * 16 * SB_STRIDE]; // 130 KiB weights
  __shared__ __align__(16) unsigned short sH[16 * SB_STRIDE];     // 16.3 KiB h(t-1)
  __shared__ float gatesS[4][16][32];                             // 8 KiB
  __shared__ float cS[16][32];                                    // 2 KiB

  const int g    = blockIdx.x;
  const int tid  = threadIdx.x;
  const int wv   = tid >> 5;
  const int lane = tid & 31;
  const int hi   = lane >> 4;
  const int nl   = lane & 15;
  const int q    = wv >> 1;
  const int half = wv & 1;
  const int nglob = q * 512 + g * 32 + half * 16 + nl;

  // ---- Preload this WGP's recurrent-weight slice into LDS ----------------
#if USE_TDM
  // 4 Tensor-DMA loads (one per gate): 32 rows x 512 bf16, padded on the fly
  // to the 520-element LDS row stride (pad_interval=256 dwords, amount=4).
  if (wv == 0) {
    const unsigned sbBase = (unsigned)(size_t)(void*)sB;
    for (int qq = 0; qq < 4; ++qq) {
      unsigned long long ga =
          (unsigned long long)(size_t)(uW + ((size_t)qq * 512 + g * 32) * LSTM_H);
      unsigned la = sbBase + (unsigned)(qq * 32 * SB_STRIDE * 2);
      v4u g0;
      g0[0] = 1u;                                   // count=1, user mode
      g0[1] = la;                                   // lds_addr
      g0[2] = (unsigned)(ga & 0xffffffffu);         // global_addr[31:0]
      g0[3] = (unsigned)((ga >> 32) & 0x01ffffffu)  // global_addr[56:32]
              | 0x80000000u;                        // type=2 ("image")
      v8i g1;
      g1[0] = (int)((1u << 16)      // data_size = 2 bytes
                  | (1u << 20)      // pad_enable
                  | (7u << 22)      // pad_interval = 256 dwords (one row)
                  | (3u << 25));    // pad_amount   = 4 dwords (8 bf16)
      g1[1] = (int)(512u << 16);    // tensor_dim0 = 512 (lo16)
      g1[2] = (int)(32u << 16);     // tensor_dim0 hi=0 | tensor_dim1 = 32
      g1[3] = (int)(512u << 16);    // tensor_dim1 hi=0 | tile_dim0 = 512
      g1[4] = 32;                   // tile_dim1 = 32 | tile_dim2 = 0
      g1[5] = 512;                  // tensor_dim0_stride lo32
      g1[6] = 0;                    // stride0 hi | stride1 lo
      g1[7] = 0;                    // stride1 hi
      v4i z4 = {0, 0, 0, 0};
#if __clang_major__ >= 23
      v8i z8 = {0, 0, 0, 0, 0, 0, 0, 0};
      __builtin_amdgcn_tensor_load_to_lds(g0, g1, z4, z4, z8, 0);
#else
      __builtin_amdgcn_tensor_load_to_lds(g0, g1, z4, z4, 0);
#endif
    }
    __builtin_amdgcn_s_wait_tensorcnt(0);
  }
#else
  for (int idx = tid; idx < 8 * 16 * 64; idx += 256) {
    int w  = idx >> 10;
    int rm = idx & 1023;
    int r  = rm >> 6;
    int kc = rm & 63;
    int nrow = (w >> 1) * 512 + g * 32 + (w & 1) * 16 + r;
    v4u d = *(const v4u*)(uW + (size_t)nrow * LSTM_H + kc * 8);
    *(v4u*)(sB + (w * 16 + r) * SB_STRIDE + kc * 8) = d;
  }
#endif
  // Zero cell state + our columns of the h broadcast buffer (h0 = c0 = 0).
  for (int idx = tid; idx < 16 * 32; idx += 256) {
    int m = idx >> 5, col = idx & 31;
    cS[m][col] = 0.0f;
    hcast[m * LSTM_H + g * 32 + col] = 0;
  }
  __syncthreads();
  __threadfence();
  __builtin_amdgcn_s_cluster_barrier();
  __threadfence();   // acquire: don't read stale hcast lines

  const float ubn = ub[nglob];
  const float* wxcol = wx + nglob;

  for (int t = 0; t < T; ++t) {
    // ---- stage h(t-1) into LDS once per step (16 KB) ---------------------
#if USE_ASYNC_STAGE
    // Multicast: all 16 WGs request the same lines; one L2 read is broadcast
    // into every WGP's LDS.  Tracked by ASYNCcnt.  Downgrades to a plain
    // global->LDS async load when not dispatched as a cluster.
    #pragma unroll
    for (int e = 0; e < 4; ++e) {
      int idx = tid + e * 256;
      int r = idx >> 6, c = idx & 63;
      unsigned ldsa = (unsigned)(size_t)(void*)(sH + r * SB_STRIDE + c * 8);
      const unsigned short* gp = hcast + r * LSTM_H + c * 8;
      asm volatile("s_mov_b32 m0, 0xffff\n\t"
                   "cluster_load_async_to_lds_b128 %0, %1, off"
                   :: "v"(ldsa), "v"(gp) : "memory");
    }
    asm volatile("s_wait_asynccnt 0x0" ::: "memory");
#else
    for (int idx = tid; idx < 16 * 64; idx += 256) {
      int r = idx >> 6, c = idx & 63;
      v4u d = *(const v4u*)(hcast + r * LSTM_H + c * 8);
      *(v4u*)(sH + r * SB_STRIDE + c * 8) = d;
    }
#endif
    __syncthreads();                       // copy done (hcast reads retired)
    // cluster-wide: everyone finished consuming h(t-1); safe to overwrite.
    __builtin_amdgcn_s_cluster_barrier();

    // ---- gates = wx_t + ub + h(t-1) @ uW^T (rows 8..15 = batch pad) ------
    v8f acc0 = {}, acc1 = {};
    if (hi == 0) {
      #pragma unroll
      for (int r = 0; r < 8; ++r)
        acc0[r] = wxcol[((size_t)r * T + t) * LSTM_G] + ubn;
      if (t + 1 < T)
        __builtin_prefetch(wxcol + ((size_t)(t + 1)) * LSTM_G, 0, 0);
    }
    const unsigned short* brow = sB + (wv * 16 + nl) * SB_STRIDE;
    const unsigned short* ah   = sH + nl * SB_STRIDE;
    #pragma unroll
    for (int kt = 0; kt < 16; kt += 2) {
      V16 a0, b0, a1, b1;
      const unsigned short* ap0 = ah + kt * 32;
      const unsigned short* bp0 = brow + kt * 32 + hi * 8;
      a0.u[0] = *(const v4u*)(ap0 + hi * 8);
      a0.u[1] = *(const v4u*)(ap0 + 16 + hi * 8);
      b0.u[0] = *(const v4u*)(bp0);
      b0.u[1] = *(const v4u*)(bp0 + 16);
      acc0 = __builtin_amdgcn_wmma_f32_16x16x32_bf16(false, a0.v, false, b0.v,
                                                     (short)0, acc0, false, false);
      const unsigned short* ap1 = ap0 + 32;
      const unsigned short* bp1 = bp0 + 32;
      a1.u[0] = *(const v4u*)(ap1 + hi * 8);
      a1.u[1] = *(const v4u*)(ap1 + 16 + hi * 8);
      b1.u[0] = *(const v4u*)(bp1);
      b1.u[1] = *(const v4u*)(bp1 + 16);
      acc1 = __builtin_amdgcn_wmma_f32_16x16x32_bf16(false, a1.v, false, b1.v,
                                                     (short)0, acc1, false, false);
    }
    #pragma unroll
    for (int r = 0; r < 8; ++r)
      gatesS[q][hi * 8 + r][half * 16 + nl] = acc0[r] + acc1[r];
    __syncthreads();

    // ---- elementwise gate math, publish h(t) -----------------------------
    #pragma unroll
    for (int e = 0; e < 2; ++e) {
      int idx = tid + e * 256;
      int m = idx >> 5, col = idx & 31;
      float i_ = sigmoidf_(gatesS[0][m][col]);
      float f_ = sigmoidf_(gatesS[1][m][col]);
      float g_ = tanhf(gatesS[2][m][col]);
      float o_ = sigmoidf_(gatesS[3][m][col]);
      float c  = f_ * cS[m][col] + i_ * g_;
      cS[m][col] = c;
      float h  = o_ * fmaxf(c, 0.0f);        // custom relu cell nonlinearity
      unsigned short hb = f2bf(h);
      hcast[m * LSTM_H + g * 32 + col] = hb;
      if (m < 8) {
        size_t o = ((size_t)m * T + t) * LSTM_H + g * 32 + col;
        if (hseq_bf)  hseq_bf[o]  = hb;
        if (hseq_f32) hseq_f32[o] = h;
        if (t == T - 1) {
          h_last[m * LSTM_H + g * 32 + col] = h;
          c_last[m * LSTM_H + g * 32 + col] = c;
        }
      }
    }
    __threadfence();          // release h(t)
    __syncthreads();
    __builtin_amdgcn_s_cluster_barrier();
    __threadfence();          // acquire before next step's staging reads
  }
}

extern "C" void kernel_launch(void* const* d_in, const int* in_sizes, int n_in,
                              void* d_out, int out_size, void* d_ws, size_t ws_size,
                              hipStream_t stream) {
  (void)in_sizes; (void)n_in; (void)out_size; (void)ws_size;
  constexpr int B = 8, T = 2048, D = 256, H = 512, G = 4 * H;

  const float* x   = (const float*)d_in[0];
  const float* w0w = (const float*)d_in[1];
  const float* w0b = (const float*)d_in[2];
  const float* u0w = (const float*)d_in[3];
  const float* u0b = (const float*)d_in[4];
  const float* w1w = (const float*)d_in[5];
  const float* w1b = (const float*)d_in[6];
  const float* u1w = (const float*)d_in[7];
  const float* u1b = (const float*)d_in[8];
  float* out = (float*)d_out;

  // workspace carve-up (~167 MB)
  size_t off = 0;
  auto alloc = [&](size_t bytes) {
    void* p = (char*)d_ws + off;
    off += (bytes + 255) & ~(size_t)255;
    return p;
  };
  unsigned short* xbf   = (unsigned short*)alloc((size_t)B * T * D * 2);
  unsigned short* w0bf  = (unsigned short*)alloc((size_t)G * D * 2);
  unsigned short* u0bf  = (unsigned short*)alloc((size_t)G * H * 2);
  unsigned short* w1bf  = (unsigned short*)alloc((size_t)G * H * 2);
  unsigned short* u1bf  = (unsigned short*)alloc((size_t)G * H * 2);
  float*          wx    = (float*)alloc((size_t)B * T * G * 4);
  unsigned short* h1bf  = (unsigned short*)alloc((size_t)B * T * H * 2);
  unsigned short* hcast = (unsigned short*)alloc((size_t)16 * H * 2);

  float* h2  = out;                          // [8, T, 512]
  float* hh0 = out + (size_t)B * T * H;      // hh[0]
  float* hh1 = hh0 + (size_t)B * H;          // hh[1]
  float* cc0 = hh0 + (size_t)2 * B * H;      // cc[0]
  float* cc1 = cc0 + (size_t)B * H;          // cc[1]

  auto cvt = [&](const float* src, unsigned short* dst, int n) {
    cvt_f32_bf16<<<(n + 255) / 256, 256, 0, stream>>>(src, dst, n);
  };
  cvt(x,   xbf,  B * T * D);
  cvt(w0w, w0bf, G * D);
  cvt(u0w, u0bf, G * H);
  cvt(w1w, w1bf, G * H);
  cvt(u1w, u1bf, G * H);

  const int M = B * T;                            // 16384 rows
  const int gemmBlocks = (M / 16) * (G / 64) / 8; // 16x64 strip per wave

  // layer 0
  wmma_gemm_bias<<<gemmBlocks, 256, 0, stream>>>(xbf, w0bf, w0b, wx, M, G, D);
  lstm_scan<<<16, 256, 0, stream>>>(wx, u0bf, u0b, hcast,
                                    h1bf, nullptr, hh0, cc0, T);
  // layer 1
  wmma_gemm_bias<<<gemmBlocks, 256, 0, stream>>>(h1bf, w1bf, w1b, wx, M, G, H);
  lstm_scan<<<16, 256, 0, stream>>>(wx, u1bf, u1b, hcast,
                                    nullptr, h2, hh1, cc1, T);
}
